// block_69028714381928
// MI455X (gfx1250) — matrix operand order repaired and checked by reading, not compiled, not run on hardware
//
#include <hip/hip_runtime.h>

// ---------------------------------------------------------------------------
// Transformer block for MI455X (gfx1250): fp32 in/out, bf16 WMMA GEMM cores.
// B=2, T=2048, D=1024, H=16, DH=64, F=4096
// All WMMA fragments are assembled from contiguous 16-byte LDS/global loads
// (ds_load_b128 / global_load_b128) -- no elementwise gathers.
// ---------------------------------------------------------------------------

#define Bc   2
#define Tc   2048
#define Dc   1024
#define Hc   16
#define DHc  64
#define Fc   4096
#define BTc  (Bc * Tc)
#define EPSc 1e-5f

typedef __attribute__((ext_vector_type(16))) __bf16 v16bf;
typedef __attribute__((ext_vector_type(8)))  __bf16 v8bf;
typedef __attribute__((ext_vector_type(8)))  float  v8f;

// D = A(16x32 bf16) * B(32x16 bf16) + C(16x16 f32)
__device__ __forceinline__ v8f wmma_bf16(v16bf a, v16bf b, v8f c) {
    return __builtin_amdgcn_wmma_f32_16x16x32_bf16(
        false, a, false, b, (short)0, c, false, false);
}

__device__ __forceinline__ v16bf concat8(v8bf lo, v8bf hi) {
    return __builtin_shufflevector(lo, hi,
        0, 1, 2, 3, 4, 5, 6, 7, 8, 9, 10, 11, 12, 13, 14, 15);
}

__device__ __forceinline__ float half16_max(float v) {
    v = fmaxf(v, __shfl_xor(v, 1, 32));
    v = fmaxf(v, __shfl_xor(v, 2, 32));
    v = fmaxf(v, __shfl_xor(v, 4, 32));
    v = fmaxf(v, __shfl_xor(v, 8, 32));
    return v;
}
__device__ __forceinline__ float half16_sum(float v) {
    v += __shfl_xor(v, 1, 32);
    v += __shfl_xor(v, 2, 32);
    v += __shfl_xor(v, 4, 32);
    v += __shfl_xor(v, 8, 32);
    return v;
}

// ---------------------------------------------------------------------------
// 128x64-tile GEMM core, double-buffered LDS, fragment-order B staging.
//   A: fp32 or bf16 row-major [M,K] (lda) ; W: fp32 row-major [K,N] (ldb)
//   8 waves, wave w owns rows [16w,16w+16), 4 N-subtiles of 16.
// A-fragment layout in LDS : row-major, padded stride 40 elems (80 B,
//   16B-aligned chunks, bank starts 20*ln mod 64 all distinct).
// B-fragment layout in LDS : per (nt,lane) contiguous 16 elems, stride 24
//   (48 B -> 16B-aligned), staged as packed uint2.
// ---------------------------------------------------------------------------
template <bool A_BF16, bool RELU, bool OUT_BF16, bool HAS_BIAS>
__device__ __forceinline__ void gemm_block(
    const float* __restrict__ Af32, const __bf16* __restrict__ Abf, int lda,
    const float* __restrict__ Bw, int ldb,
    const float* __restrict__ bias,
    float* __restrict__ Of32, __bf16* __restrict__ Obf, int ldo, int out_trans,
    int m0, int n0, int K)
{
    constexpr int LDA = 40;   // padded A row stride (elems)
    constexpr int BCH = 24;   // B fragment chunk stride (elems)
    __shared__ __attribute__((aligned(16))) __bf16 As[2][128 * LDA];
    __shared__ __attribute__((aligned(16))) __bf16 BsF[2][4 * 32 * BCH];

    const int tid  = threadIdx.x;
    const int wave = tid >> 5;
    const int lane = tid & 31;
    const int half = lane >> 4;
    const int ln   = lane & 15;

    v8f acc[4] = {};
    const int nk = K >> 5;

    auto stage = [&](int k0, int buf) {
        // ---- A tile 128x32 -> bf16, row-major stride 40 ----
        #pragma unroll
        for (int it = 0; it < 4; ++it) {
            int idx = tid + it * 256;            // 1024 groups of 4 elems
            int r = idx >> 3, c = (idx & 7) * 4;
            union { uint2 u; __bf16 h[4]; } t;
            if (A_BF16) {
                t.u = *(const uint2*)(Abf + (long)(m0 + r) * lda + k0 + c);
            } else {
                float4 v = *(const float4*)(Af32 + (long)(m0 + r) * lda + k0 + c);
                t.h[0] = (__bf16)v.x; t.h[1] = (__bf16)v.y;
                t.h[2] = (__bf16)v.z; t.h[3] = (__bf16)v.w;
            }
            *(uint2*)&As[buf][r * LDA + c] = t.u;
        }
        // ---- W tile 32x64 -> bf16 in fragment order ----
        // group g: ln (col-in-subtile, fast for coalescing), half, ig, nt
        #pragma unroll
        for (int it = 0; it < 2; ++it) {
            int g = tid + it * 256;              // 512 groups of 4 elems
            int nl = g & 15, rest = g >> 4;
            int hf = rest & 1, ig = (rest >> 1) & 3, ntb = rest >> 3;
            int gcol = n0 + ntb * 16 + nl;
            union { uint2 u; __bf16 h[4]; } t;
            #pragma unroll
            for (int j = 0; j < 4; ++j) {
                int gk = k0 + 16 * hf + ig * 4 + j;
                t.h[j] = (__bf16)Bw[(long)gk * ldb + gcol];
            }
            *(uint2*)&BsF[buf][(ntb * 32 + nl + 16 * hf) * BCH + ig * 4] = t.u;
        }
    };

    stage(0, 0);
    __syncthreads();

    for (int kt = 0; kt < nk; ++kt) {
        int buf = kt & 1;
        if (kt + 1 < nk) stage((kt + 1) << 5, buf ^ 1);

        // A fragment: two contiguous 16B chunks (contiguous k = 16*half + s)
        const __bf16* Ab = &As[buf][(16 * wave + ln) * LDA];
        v16bf a = concat8(*(const v8bf*)(Ab + 16 * half),
                          *(const v8bf*)(Ab + 16 * half + 8));
        #pragma unroll
        for (int nt = 0; nt < 4; ++nt) {
            const __bf16* Bb = &BsF[buf][(nt * 32 + lane) * BCH];
            v16bf b = concat8(*(const v8bf*)Bb, *(const v8bf*)(Bb + 8));
            acc[nt] = wmma_bf16(a, b, acc[nt]);
        }
        __syncthreads();
    }

    // ---- epilogue ----
    #pragma unroll
    for (int nt = 0; nt < 4; ++nt) {
        int gn = n0 + nt * 16 + ln;
        float bi = HAS_BIAS ? bias[gn] : 0.0f;
        if (OUT_BF16 && out_trans) {
            // transposed store: 8 row-consecutive results -> one 16B store
            union { uint4 u; __bf16 h[8]; } t;
            #pragma unroll
            for (int r = 0; r < 8; ++r) {
                float v = acc[nt][r];
                if (HAS_BIAS) v += bi;
                if (RELU) v = fmaxf(v, 0.0f);
                t.h[r] = (__bf16)v;
            }
            *(uint4*)&Obf[(long)gn * ldo + m0 + 16 * wave + 8 * half] = t.u;
        } else {
            #pragma unroll
            for (int r = 0; r < 8; ++r) {
                int gm = m0 + 16 * wave + r + 8 * half;
                float v = acc[nt][r];
                if (HAS_BIAS) v += bi;
                if (RELU) v = fmaxf(v, 0.0f);
                if (OUT_BF16) Obf[(long)gm * ldo + gn] = (__bf16)v;
                else          Of32[(long)gm * ldo + gn] = v;
            }
        }
    }
}

// ---------------------------------------------------------------------------
// Kernel 1: fused QKV projection.  grid = (T/128, 3*H, B)
//   q,k stored [B,H,T,DH]; v stored TRANSPOSED [B,H,DH,T] so the attention
//   P*V B-fragments become contiguous-in-key loads.
// ---------------------------------------------------------------------------
__global__ __launch_bounds__(256) void qkv_kernel(
    const float* __restrict__ x,
    const float* __restrict__ Wq, const float* __restrict__ Wk,
    const float* __restrict__ Wv,
    __bf16* __restrict__ q, __bf16* __restrict__ k, __bf16* __restrict__ v)
{
    int mt = blockIdx.x, z = blockIdx.y, b = blockIdx.z;
    int which = z / Hc, h = z % Hc;
    const float* W = (which == 0 ? Wq : which == 1 ? Wk : Wv) + (long)h * Dc * DHc;
    const float* A = x + (long)b * Tc * Dc;
    if (which == 2) {
        __bf16* out = v + (long)(b * Hc + h) * DHc * Tc;   // [DH,T]
        gemm_block<false, false, true, false>(A, nullptr, Dc, W, DHc, nullptr,
                                              nullptr, out, Tc, 1,
                                              mt * 128, 0, Dc);
    } else {
        __bf16* out = (which == 0 ? q : k) + (long)(b * Hc + h) * Tc * DHc;
        gemm_block<false, false, true, false>(A, nullptr, Dc, W, DHc, nullptr,
                                              nullptr, out, DHc, 0,
                                              mt * 128, 0, Dc);
    }
}

// ---------------------------------------------------------------------------
// Kernel 2: causal flash attention.  grid = (T/128, H, B), 256 threads.
//   q,k bf16 [B,H,T,DH]; v bf16 [B,H,DH,T]; out o bf16 [B,T,D].
//   Q fragments loaded directly from global (16B contiguous per lane).
//   K/V tiles double-buffered in LDS with padded stride 72 (144 B).
// ---------------------------------------------------------------------------
__global__ __launch_bounds__(256) void attn_kernel(
    const __bf16* __restrict__ qb, const __bf16* __restrict__ kb,
    const __bf16* __restrict__ vtb, __bf16* __restrict__ ob)
{
    const float SCALE = 0.125f;                  // 1/sqrt(64)
    const float NEG_INF = -__builtin_inff();
    constexpr int LDK = 72;                      // padded row stride

    int mt = blockIdx.x, h = blockIdx.y, b = blockIdx.z;
    const __bf16* Q  = qb  + (long)(b * Hc + h) * Tc * DHc;
    const __bf16* Kp = kb  + (long)(b * Hc + h) * Tc * DHc;
    const __bf16* Vt = vtb + (long)(b * Hc + h) * DHc * Tc;
    int m0 = mt * 128;

    __shared__ __attribute__((aligned(16))) __bf16 Ks[2][64 * LDK];   // [key][e]
    __shared__ __attribute__((aligned(16))) __bf16 VsT[2][64 * LDK];  // [e][key]
    __shared__ __attribute__((aligned(16))) __bf16 Ps[8 * 16 * LDK];  // per-wave

    const int tid = threadIdx.x, wave = tid >> 5, lane = tid & 31;
    const int half = lane >> 4, ln = lane & 15;

    // Q A-fragments straight from global: row = 16*wave + ln, two 16B chunks
    v16bf aq[2];
    #pragma unroll
    for (int kk = 0; kk < 2; ++kk) {
        const __bf16* qrow = Q + (long)(m0 + 16 * wave + ln) * DHc
                               + kk * 32 + 16 * half;
        aq[kk] = concat8(*(const v8bf*)qrow, *(const v8bf*)(qrow + 8));
    }

    auto stage_kv = [&](int kt, int buf) {
        #pragma unroll
        for (int it = 0; it < 4; ++it) {
            int idx = tid + it * 256;            // 1024 groups of 4 elems
            int r = idx >> 4, c = (idx & 15) * 4;
            *(uint2*)&Ks[buf][r * LDK + c] =
                *(const uint2*)(Kp + (long)(kt + r) * DHc + c);   // K[key][e]
            *(uint2*)&VsT[buf][r * LDK + c] =
                *(const uint2*)(Vt + (long)r * Tc + kt + c);      // V^T[e][key]
        }
    };

    v8f oacc[4] = {};
    float mrun[8], lrun[8];
    #pragma unroll
    for (int r = 0; r < 8; ++r) { mrun[r] = NEG_INF; lrun[r] = 0.0f; }

    const int ntiles = m0 / 64 + 2;              // causal clip
    stage_kv(0, 0);
    __syncthreads();

    __bf16* Pw = Ps + wave * 16 * LDK;

    for (int ti = 0; ti < ntiles; ++ti) {
        int buf = ti & 1;
        int kt = ti * 64;
        if (ti + 1 < ntiles) stage_kv(kt + 64, buf ^ 1);

        // S = Q K^T : B-frag contiguous in e within K row
        v8f s[4] = {};
        #pragma unroll
        for (int kk = 0; kk < 2; ++kk) {
            #pragma unroll
            for (int nt = 0; nt < 4; ++nt) {
                const __bf16* kr = &Ks[buf][(nt * 16 + ln) * LDK
                                            + kk * 32 + 16 * half];
                v16bf bk = concat8(*(const v8bf*)kr, *(const v8bf*)(kr + 8));
                s[nt] = wmma_bf16(aq[kk], bk, s[nt]);
            }
        }

        // online softmax; row stats replicated across each 16-lane half
        #pragma unroll
        for (int r = 0; r < 8; ++r) {
            int gq = m0 + 16 * wave + r + 8 * half;
            float pv[4];
            float rowmax = NEG_INF;
            #pragma unroll
            for (int nt = 0; nt < 4; ++nt) {
                int key = kt + nt * 16 + ln;
                float sv = (key <= gq) ? s[nt][r] * SCALE : NEG_INF;
                pv[nt] = sv;
                rowmax = fmaxf(rowmax, sv);
            }
            rowmax = half16_max(rowmax);
            float mnew = fmaxf(mrun[r], rowmax);
            float rowsum = 0.0f;
            #pragma unroll
            for (int nt = 0; nt < 4; ++nt) {
                float p = __expf(pv[nt] - mnew);   // exp(-inf)=0 when masked
                s[nt][r] = p;
                rowsum += p;
            }
            rowsum = half16_sum(rowsum);
            float alpha = __expf(mrun[r] - mnew);
            lrun[r] = lrun[r] * alpha + rowsum;
            mrun[r] = mnew;
            #pragma unroll
            for (int nt = 0; nt < 4; ++nt) oacc[nt][r] *= alpha;
        }

        // restage P (C-layout -> A-layout); per-wave strip, wave-local only
        #pragma unroll
        for (int nt = 0; nt < 4; ++nt)
            #pragma unroll
            for (int r = 0; r < 8; ++r)
                Pw[(r + 8 * half) * LDK + nt * 16 + ln] = (__bf16)s[nt][r];

        // O += P * V : A-frag from Ps (two 16B chunks), B-frag from V^T
        #pragma unroll
        for (int kk = 0; kk < 2; ++kk) {
            const __bf16* pr = Pw + ln * LDK + kk * 32 + 16 * half;
            v16bf ap = concat8(*(const v8bf*)pr, *(const v8bf*)(pr + 8));
            #pragma unroll
            for (int nt = 0; nt < 4; ++nt) {
                const __bf16* vr = &VsT[buf][(nt * 16 + ln) * LDK
                                             + kk * 32 + 16 * half];
                v16bf bv = concat8(*(const v8bf*)vr, *(const v8bf*)(vr + 8));
                oacc[nt] = wmma_bf16(ap, bv, oacc[nt]);
            }
        }
        __syncthreads();
    }

    // normalize + store head-concat [B,T,D]
    #pragma unroll
    for (int nt = 0; nt < 4; ++nt) {
        #pragma unroll
        for (int r = 0; r < 8; ++r) {
            int t = m0 + 16 * wave + r + 8 * half;
            int d = h * DHc + nt * 16 + ln;
            ob[((long)(b * Tc + t)) * Dc + d] = (__bf16)(oacc[nt][r] / lrun[r]);
        }
    }
}

// ---------------------------------------------------------------------------
// GEMM wrappers
// ---------------------------------------------------------------------------
__global__ __launch_bounds__(256) void oproj_kernel(
    const __bf16* __restrict__ o, const float* __restrict__ Wo,
    const float* __restrict__ bo, float* __restrict__ y)
{
    gemm_block<true, false, false, true>(nullptr, o, Dc, Wo, Dc, bo,
                                         y, nullptr, Dc, 0,
                                         blockIdx.x * 128, blockIdx.y * 64, Dc);
}

__global__ __launch_bounds__(256) void ffn1_kernel(
    const __bf16* __restrict__ a, const float* __restrict__ W1,
    const float* __restrict__ b1, __bf16* __restrict__ mid)
{
    gemm_block<true, true, true, true>(nullptr, a, Dc, W1, Fc, b1,
                                       nullptr, mid, Fc, 0,
                                       blockIdx.x * 128, blockIdx.y * 64, Dc);
}

__global__ __launch_bounds__(256) void ffn2_kernel(
    const __bf16* __restrict__ mid, const float* __restrict__ W2,
    const float* __restrict__ b2, float* __restrict__ f)
{
    gemm_block<true, false, false, true>(nullptr, mid, Fc, W2, Dc, b2,
                                         f, nullptr, Dc, 0,
                                         blockIdx.x * 128, blockIdx.y * 64, Fc);
}

// ---------------------------------------------------------------------------
// Residual + LayerNorm, one row (D=1024) per 256-thread block.
// OUT_BF16 selects bf16 (FFN input) vs fp32 (final output) store.
// ---------------------------------------------------------------------------
__device__ __forceinline__ float block_sum(float v, float* red) {
    v += __shfl_xor(v, 1, 32);  v += __shfl_xor(v, 2, 32);
    v += __shfl_xor(v, 4, 32);  v += __shfl_xor(v, 8, 32);
    v += __shfl_xor(v, 16, 32);
    int wave = threadIdx.x >> 5, lane = threadIdx.x & 31;
    if (lane == 0) red[wave] = v;
    __syncthreads();
    float r = 0.0f;
    #pragma unroll
    for (int i = 0; i < 8; ++i) r += red[i];
    __syncthreads();
    return r;
}

template <bool OUT_BF16>
__device__ __forceinline__ void ln_residual_body(
    const float* __restrict__ y, const float* __restrict__ x,
    const float* __restrict__ g, const float* __restrict__ be,
    float* __restrict__ out_f32, __bf16* __restrict__ out_bf)
{
    __shared__ float red[8];
    long row = blockIdx.x;
    const float* yr = y + row * Dc;
    const float* xr = x + row * Dc;

    float vals[4];
    float s = 0.0f;
    #pragma unroll
    for (int i = 0; i < 4; ++i) {
        vals[i] = yr[threadIdx.x + i * 256];
        s += vals[i];
    }
    float mu = block_sum(s, red) * (1.0f / Dc);
    float vs = 0.0f;
    #pragma unroll
    for (int i = 0; i < 4; ++i) {
        float d = vals[i] - mu;
        vs += d * d;
    }
    float rstd = rsqrtf(block_sum(vs, red) * (1.0f / Dc) + EPSc);
    #pragma unroll
    for (int i = 0; i < 4; ++i) {
        int c = threadIdx.x + i * 256;
        float r = xr[c] + (vals[i] - mu) * rstd * g[c] + be[c];
        if (OUT_BF16) out_bf[row * Dc + c] = (__bf16)r;
        else          out_f32[row * Dc + c] = r;
    }
}

__global__ __launch_bounds__(256) void ln_residual_bf16_kernel(
    const float* __restrict__ y, const float* __restrict__ x,
    const float* __restrict__ g, const float* __restrict__ be,
    __bf16* __restrict__ out_bf)
{
    ln_residual_body<true>(y, x, g, be, nullptr, out_bf);
}

__global__ __launch_bounds__(256) void ln_residual_f32_kernel(
    const float* __restrict__ y, const float* __restrict__ x,
    const float* __restrict__ g, const float* __restrict__ be,
    float* __restrict__ out_f32)
{
    ln_residual_body<false>(y, x, g, be, out_f32, nullptr);
}

// ---------------------------------------------------------------------------
// Host-side launch chain
// ---------------------------------------------------------------------------
extern "C" void kernel_launch(void* const* d_in, const int* in_sizes, int n_in,
                              void* d_out, int out_size, void* d_ws, size_t ws_size,
                              hipStream_t stream)
{
    const float* emb = (const float*)d_in[0];
    const float* Wq  = (const float*)d_in[1];
    const float* Wk  = (const float*)d_in[2];
    const float* Wv  = (const float*)d_in[3];
    const float* Wo  = (const float*)d_in[4];
    const float* bo  = (const float*)d_in[5];
    const float* W1  = (const float*)d_in[6];
    const float* b1  = (const float*)d_in[7];
    const float* W2  = (const float*)d_in[8];
    const float* b2  = (const float*)d_in[9];
    const float* g1  = (const float*)d_in[10];
    const float* be1 = (const float*)d_in[11];
    const float* g2  = (const float*)d_in[12];
    const float* be2 = (const float*)d_in[13];
    float* out = (float*)d_out;

    // workspace layout:
    //   q,k : [B,H,T,DH] bf16 ; v : [B,H,DH,T] bf16 ; o : [B,T,D] bf16
    //   s32 : [B*T,D] fp32 scratch (attn_out, reused for ffn out)
    //   a_bf: [B*T,D] bf16 ; mid: [B*T,F] bf16          (total ~88 MiB)
    const size_t NQKV = (size_t)Bc * Hc * Tc * DHc;     // == B*T*D
    __bf16* qb   = (__bf16*)d_ws;
    __bf16* kbuf = qb + NQKV;
    __bf16* vbuf = kbuf + NQKV;
    __bf16* obuf = vbuf + NQKV;
    float*  s32  = (float*)(obuf + NQKV);
    __bf16* a_bf = (__bf16*)(s32 + (size_t)BTc * Dc);
    __bf16* mid  = a_bf + (size_t)BTc * Dc;

    dim3 blk(256);

    qkv_kernel<<<dim3(Tc / 128, 3 * Hc, Bc), blk, 0, stream>>>(
        emb, Wq, Wk, Wv, qb, kbuf, vbuf);

    attn_kernel<<<dim3(Tc / 128, Hc, Bc), blk, 0, stream>>>(qb, kbuf, vbuf, obuf);

    oproj_kernel<<<dim3(BTc / 128, Dc / 64), blk, 0, stream>>>(obuf, Wo, bo, s32);

    ln_residual_bf16_kernel<<<dim3(BTc), blk, 0, stream>>>(
        s32, emb, g1, be1, a_bf);

    ffn1_kernel<<<dim3(BTc / 128, Fc / 64), blk, 0, stream>>>(a_bf, W1, b1, mid);

    ffn2_kernel<<<dim3(BTc / 128, Dc / 64), blk, 0, stream>>>(mid, W2, b2, s32);

    ln_residual_f32_kernel<<<dim3(BTc), blk, 0, stream>>>(
        s32, emb, g2, be2, out);
}
